// GaussianSplatScene_68856915689837
// MI455X (gfx1250) — matrix operand order repaired and checked by reading, not compiled, not run on hardware
//
#include <hip/hip_runtime.h>

typedef __attribute__((ext_vector_type(2)))  float    v2f;
typedef __attribute__((ext_vector_type(8)))  float    v8f;
typedef __attribute__((ext_vector_type(8)))  _Float16 v8h;
typedef __attribute__((ext_vector_type(16))) _Float16 v16h;

#define WAVES_PER_BLOCK 4
#define PROW 40   // LDS row stride in halves (80B) -> conflict-free banks, 16B-aligned v8h

// exp(-0.5*x) = exp2(x * -0.5*log2(e)); constant folded into the query-side WMMA operand
#define NEG_HALF_LOG2E (-0.72134752044448170368f)

// ---------------------------------------------------------------------------
// Kernel 1: per-gaussian descriptor v[12] = [A00,A11,A22,2A01,2A02,2A12,
//   -2*(A mu), muAmu - 2*ln sigmoid(op), 0, 0], written directly in the
//   f32 WMMA A-operand layout:
//   subtile st = n/16 (16 gaussians), element (m=n%16, k):
//     lane = m + 16*((k%4)/2), vgpr = k%2, wmma j = k/4
//   flat f32 index = ((st*3 + j)*32 + lane)*2 + vgpr
// ---------------------------------------------------------------------------
__global__ void gs_precompute(const float* __restrict__ means,
                              const float* __restrict__ log_scales,
                              const float* __restrict__ rot,
                              const float* __restrict__ opac,
                              float* __restrict__ gvecA, int N) {
  int n = blockIdx.x * blockDim.x + threadIdx.x;
  if (n >= N) return;
  float qw = rot[n*4+0], qx = rot[n*4+1], qy = rot[n*4+2], qz = rot[n*4+3];
  float nr = fmaxf(sqrtf(qw*qw + qx*qx + qy*qy + qz*qz), 1e-12f);
  float iv = 1.0f / nr;
  qw *= iv; qx *= iv; qy *= iv; qz *= iv;
  float r00 = 1.f - 2.f*(qy*qy + qz*qz), r01 = 2.f*(qx*qy - qw*qz), r02 = 2.f*(qx*qz + qw*qy);
  float r10 = 2.f*(qx*qy + qw*qz), r11 = 1.f - 2.f*(qx*qx + qz*qz), r12 = 2.f*(qy*qz - qw*qx);
  float r20 = 2.f*(qx*qz - qw*qy), r21 = 2.f*(qy*qz + qw*qx), r22 = 1.f - 2.f*(qx*qx + qy*qy);
  float s0 = __expf(log_scales[n*3+0]);
  float s1 = __expf(log_scales[n*3+1]);
  float s2 = __expf(log_scales[n*3+2]);
  float m00 = r00*s0, m01 = r01*s1, m02 = r02*s2;
  float m10 = r10*s0, m11 = r11*s1, m12 = r12*s2;
  float m20 = r20*s0, m21 = r21*s1, m22 = r22*s2;
  float c00 = m00*m00 + m01*m01 + m02*m02 + 1e-6f;
  float c01 = m00*m10 + m01*m11 + m02*m12;
  float c02 = m00*m20 + m01*m21 + m02*m22;
  float c11 = m10*m10 + m11*m11 + m12*m12 + 1e-6f;
  float c12 = m10*m20 + m11*m21 + m12*m22;
  float c22 = m20*m20 + m21*m21 + m22*m22 + 1e-6f;
  float d0 = c11*c22 - c12*c12;
  float d1 = c02*c12 - c01*c22;
  float d2 = c01*c12 - c02*c11;
  float det = c00*d0 + c01*d1 + c02*d2;
  float id  = 1.0f / det;
  float a00 = d0*id, a01 = d1*id, a02 = d2*id;
  float a11 = (c00*c22 - c02*c02)*id;
  float a12 = (c01*c02 - c00*c12)*id;
  float a22 = (c00*c11 - c01*c01)*id;
  float mx = means[n*3+0], my = means[n*3+1], mz = means[n*3+2];
  float am0 = a00*mx + a01*my + a02*mz;
  float am1 = a01*mx + a11*my + a12*mz;
  float am2 = a02*mx + a12*my + a22*mz;
  float muAmu = mx*am0 + my*am1 + mz*am2;
  float op = opac[n];
  float lnop = -log1pf(__expf(-op));           // ln sigmoid(op)

  float vals[12];
  vals[0] = a00; vals[1] = a11; vals[2] = a22;
  vals[3] = 2.f*a01; vals[4] = 2.f*a02; vals[5] = 2.f*a12;
  vals[6] = -2.f*am0; vals[7] = -2.f*am1; vals[8] = -2.f*am2;
  vals[9] = muAmu - 2.f*lnop; vals[10] = 0.f; vals[11] = 0.f;

  const int st = n >> 4;
  const int m  = n & 15;
#pragma unroll
  for (int k = 0; k < 12; k++) {
    int j    = k >> 2;
    int kr   = k & 3;
    int lane = m + 16 * (kr >> 1);
    int v    = kr & 1;
    gvecA[((st*3 + j)*32 + lane)*2 + v] = vals[k];
  }
}

// ---------------------------------------------------------------------------
// Kernel 2: shuffle feats (N x 32 f32) into the f16 WMMA A-operand layout.
//  A = feats^T tile (16 f-rows x 32 n-cols), element (m,k):
//    lane = m + 16*((k%16)/8), vgpr = ((k%8)/2) + 4*(k/16), half = k%2
//  flat f16 index = c*1024 + ftile*512 + lane*16 + vgpr*2 + half
// ---------------------------------------------------------------------------
__global__ void gs_shuffle_feats(const float* __restrict__ feats,
                                 _Float16* __restrict__ featsA, int total) {
  int tid = blockIdx.x * blockDim.x + threadIdx.x;
  if (tid >= total) return;
  int h  =  tid        & 1;
  int v  = (tid >> 1)  & 7;
  int l  = (tid >> 4)  & 31;
  int ft = (tid >> 9)  & 1;
  int c  =  tid >> 10;
  int k  = 16*(v >> 2) + 8*(l >> 4) + 2*(v & 3) + h;   // n within chunk
  int m  = l & 15;                                     // f within tile
  int n  = 32*c + k;
  int f  = 16*ft + m;
  featsA[tid] = (_Float16)feats[n*32 + f];
}

// ---------------------------------------------------------------------------
// Kernel 3: fused main loop.  One wave per 16-query tile; flash-style loop
// over 32-gaussian chunks.  S^T via f32 WMMA (K=12, exp2 scale pre-folded
// into the query operand), P = v_exp_f32 on TRANS pipe, O^T and row-sums
// accumulated via f16 WMMA.
// ---------------------------------------------------------------------------
__global__ void __launch_bounds__(32 * WAVES_PER_BLOCK)
gs_main(const float* __restrict__ query,
        const float* __restrict__ gvecA,
        const _Float16* __restrict__ featsA,
        float* __restrict__ out, int N) {
  __shared__ alignas(32) _Float16 Pbuf[WAVES_PER_BLOCK][16][PROW];
  const int lane = threadIdx.x & 31;
  const int w    = threadIdx.x >> 5;
  const int tile = blockIdx.x * WAVES_PER_BLOCK + w;
  const int qy   = lane & 15;
  const int hi   = lane >> 4;
  const int qg   = tile * 16 + qy;

  // query-side u vector, pre-scaled by -0.5*log2(e) so the f32 WMMA chain
  // directly yields the exp2 argument (no per-chunk scaling VALU).
  const float C = NEG_HALF_LOG2E;
  float x = query[qg*3+0], y = query[qg*3+1], z = query[qg*3+2];
  float u0 = C*x*x, u1 = C*y*y, u2 = C*z*z, u3 = C*x*y, u4 = C*x*z, u5 = C*y*z;
  v2f bq0, bq1, bq2;
  bq0.x = hi ? u2    : u0;    bq0.y = hi ? u3    : u1;
  bq1.x = hi ? C*x   : u4;    bq1.y = hi ? C*y   : u5;
  bq2.x = hi ? 0.f   : C*z;   bq2.y = hi ? 0.f   : C;

  v8f acc0 = {}, acc1 = {}, accS = {};
  v16h ones;
#pragma unroll
  for (int i = 0; i < 16; i++) ones[i] = (_Float16)1.0f;

  const v2f* gA = (const v2f*)gvecA;
  const int NC = N >> 5;            // chunks of 32 gaussians
#pragma unroll 2
  for (int c = 0; c < NC; c++) {
    const int gbase = c * 6 * 32;   // two subtiles x 3 wmma-steps x 32 lanes
#pragma unroll
    for (int s = 0; s < 2; s++) {   // two 16-gaussian subtiles
      v2f a0 = gA[gbase + (s*3 + 0)*32 + lane];
      v2f a1 = gA[gbase + (s*3 + 1)*32 + lane];
      v2f a2 = gA[gbase + (s*3 + 2)*32 + lane];
      v8f st = {};
      st = __builtin_amdgcn_wmma_f32_16x16x4_f32(false, a0, false, bq0, (short)0, st, false, false);
      st = __builtin_amdgcn_wmma_f32_16x16x4_f32(false, a1, false, bq1, (short)0, st, false, false);
      st = __builtin_amdgcn_wmma_f32_16x16x4_f32(false, a2, false, bq2, (short)0, st, false, false);
      v8h ph;
#pragma unroll
      for (int i = 0; i < 8; i++)
        ph[i] = (_Float16)__builtin_amdgcn_exp2f(st[i]);
      // S^T D-tile: lane holds col q, rows n = s*16 + hi*8 + r  -> one b128 store
      *(v8h*)&Pbuf[w][qy][s*16 + hi*8] = ph;
    }
    // B operand (P^T 32x16): lane = q + 16*hi; elems 0..7 -> n = 8*hi + i, elems 8..15 -> n = 16 + 8*hi + i
    v8h bl0 = *(const v8h*)&Pbuf[w][qy][hi*8 + 0];
    v8h bl1 = *(const v8h*)&Pbuf[w][qy][hi*8 + 16];
    v16h bp;
#pragma unroll
    for (int i = 0; i < 8; i++) { bp[i] = bl0[i]; bp[8+i] = bl1[i]; }

    const v16h* afp = (const v16h*)featsA + c*64 + lane;   // 32B per lane, coalesced
    v16h af0 = afp[0];
    v16h af1 = afp[32];
    acc0 = __builtin_amdgcn_wmma_f32_16x16x32_f16(false, af0,  false, bp, (short)0, acc0, false, false);
    acc1 = __builtin_amdgcn_wmma_f32_16x16x32_f16(false, af1,  false, bp, (short)0, acc1, false, false);
    accS = __builtin_amdgcn_wmma_f32_16x16x32_f16(false, ones, false, bp, (short)0, accS, false, false);

    // branchless next-chunk prefetch (clamped); locality 3 -> WGP-scope prefetch
    // (SYS-scope prefetch would only pull into GL2, where this already lives)
    const int cn = (c + 1 < NC) ? (c + 1) : c;
    __builtin_prefetch(gA + cn*192 + lane, 0, 3);
    __builtin_prefetch((const void*)((const v16h*)featsA + cn*64 + lane), 0, 3);
  }

  // O^T D-tile: lane holds col q, rows f = ftile*16 + hi*8 + r; accS rows all equal rowsum(q)
  float inv[8];
#pragma unroll
  for (int i = 0; i < 8; i++) inv[i] = __builtin_amdgcn_rcpf(accS[i] + 1e-8f);

  float4 o;
  float* p0 = out + qg*32 + hi*8;
  o.x = acc0[0]*inv[0]; o.y = acc0[1]*inv[1]; o.z = acc0[2]*inv[2]; o.w = acc0[3]*inv[3];
  *(float4*)p0 = o;
  o.x = acc0[4]*inv[4]; o.y = acc0[5]*inv[5]; o.z = acc0[6]*inv[6]; o.w = acc0[7]*inv[7];
  *(float4*)(p0 + 4) = o;
  float* p1 = out + qg*32 + 16 + hi*8;
  o.x = acc1[0]*inv[0]; o.y = acc1[1]*inv[1]; o.z = acc1[2]*inv[2]; o.w = acc1[3]*inv[3];
  *(float4*)p1 = o;
  o.x = acc1[4]*inv[4]; o.y = acc1[5]*inv[5]; o.z = acc1[6]*inv[6]; o.w = acc1[7]*inv[7];
  *(float4*)(p1 + 4) = o;
}

// ---------------------------------------------------------------------------
extern "C" void kernel_launch(void* const* d_in, const int* in_sizes, int n_in,
                              void* d_out, int out_size, void* d_ws, size_t ws_size,
                              hipStream_t stream) {
  const float* query  = (const float*)d_in[0];
  const float* means  = (const float*)d_in[1];
  const float* lsc    = (const float*)d_in[2];
  const float* rot    = (const float*)d_in[3];
  const float* opac   = (const float*)d_in[4];
  const float* feats  = (const float*)d_in[5];
  float* out = (float*)d_out;

  const int Q = in_sizes[0] / 3;
  const int N = in_sizes[1] / 3;

  float*    gvecA  = (float*)d_ws;                                              // N*12 f32 = 768 KB
  _Float16* featsA = (_Float16*)((char*)d_ws + (size_t)N * 12 * sizeof(float)); // N*32 f16 = 1 MB

  gs_precompute<<<(N + 255) / 256, 256, 0, stream>>>(means, lsc, rot, opac, gvecA, N);

  const int total = N * 32;
  gs_shuffle_feats<<<(total + 255) / 256, 256, 0, stream>>>(feats, featsA, total);

  const int tiles  = Q / 16;
  const int blocks = tiles / WAVES_PER_BLOCK;
  gs_main<<<blocks, 32 * WAVES_PER_BLOCK, 0, stream>>>(query, gvecA, featsA, out, N);
}
